// CCPL_7206955123015
// MI455X (gfx1250) — matrix operand order repaired and checked by reading, not compiled, not run on hardware
//
#include <hip/hip_runtime.h>

// ---------------------------------------------------------------------------
// CCPL loss for MI455X (gfx1250): bf16 WMMA GEMMs + flash-style online LSE.
// ---------------------------------------------------------------------------

typedef __bf16 bf16;
typedef __bf16 v16bf __attribute__((ext_vector_type(16)));
typedef float  v8f   __attribute__((ext_vector_type(8)));

union FragU { v16bf v; unsigned int u[8]; };

static constexpr int S_PAIRS = 4096;   // neighbor pairs per image
static constexpr int BATCH   = 4;
static constexpr int MROWS   = BATCH * S_PAIRS;  // 16384 MLP rows

// ----- WMMA fragment loaders (CDNA5 16-bit layouts, wave32) -----------------
// A 16x32 (MxK): lane&15 = row, lane>>4 selects K-half; VGPR j packs 2 K vals.
__device__ __forceinline__ v16bf load_a_frag(const bf16* __restrict__ base,
                                             int ld, int kbase, int lane) {
  FragU f;
  const int row  = lane & 15;
  const int half = lane >> 4;
  const bf16* p = base + (size_t)row * ld + kbase + 8 * half;
#pragma unroll
  for (int j = 0; j < 8; ++j) {
    const int k0 = (j < 4) ? (2 * j) : (16 + 2 * (j - 4));
    f.u[j] = *(const unsigned int*)(p + k0);
  }
  return f.v;
}

// B 32x16 (KxN) sourced from a row-major [N,K] array (W or key rows):
// lane&15 = column n, lane>>4 selects K-half; VGPR j packs 2 K vals (same order as A).
__device__ __forceinline__ v16bf load_b_frag(const bf16* __restrict__ base,
                                             int ld, int kbase, int lane) {
  FragU f;
  const int n    = lane & 15;
  const int half = lane >> 4;
  const bf16* p = base + (size_t)n * ld + kbase + 8 * half;
#pragma unroll
  for (int j = 0; j < 8; ++j)
    f.u[j] = *(const unsigned int*)(p + ((j < 4) ? (2 * j) : (16 + 2 * (j - 4))));
  return f.v;
}

__device__ __forceinline__ v8f wmma_bf16(v16bf a, v16bf b, v8f c) {
  return __builtin_amdgcn_wmma_f32_16x16x32_bf16(false, a, false, b,
                                                 (short)0, c, false, false);
}

// ----- small utility kernels ------------------------------------------------
__global__ void k_zero_f32(float* p, int n) {
  int i = blockIdx.x * blockDim.x + threadIdx.x;
  if (i < n) p[i] = 0.0f;
}

__global__ void k_zero_u32(unsigned int* p, int n) {
  int i = blockIdx.x * blockDim.x + threadIdx.x;
  if (i < n) p[i] = 0u;
}

__global__ void k_cvt_bf16(const float* __restrict__ src, bf16* __restrict__ dst, int n) {
  int i = blockIdx.x * blockDim.x + threadIdx.x;
  if (i < n) dst[i] = (bf16)src[i];
}

// gather + difference: X[b*S+s, c] = feat[b,c,cid[s]] - feat[b,c,nid[s]]
__global__ void k_gather_diff(const float* __restrict__ feat,
                              const int* __restrict__ cid,
                              const int* __restrict__ nid,
                              bf16* __restrict__ X, int C, int HW) {
  int idx = blockIdx.x * blockDim.x + threadIdx.x;
  const int total = MROWS * C;
  if (idx >= total) return;
  const int c   = idx % C;
  const int row = idx / C;                 // b*S + s
  const int s   = row & (S_PAIRS - 1);
  const int b   = row >> 12;
  const size_t base = ((size_t)b * C + c) * (size_t)HW;
  const float v = feat[base + cid[s]] - feat[base + nid[s]];
  X[(size_t)row * C + c] = (bf16)v;
}

// ----- GEMM: Y = act(X[M,K] @ W[N,K]^T + bias), bf16 in / f32 acc / bf16 out.
// Block = 4 waves; block tile = 64 rows x (16*NT) cols; wave tile 16 x 16*NT.
// 16*NT weight rows staged in LDS (row stride 520 bf16 -> conflict-free).
template <int NT>
__global__ void __launch_bounds__(128)
k_gemm_wmma(const bf16* __restrict__ X, int ldx,
            const bf16* __restrict__ W, int ldw,
            const float* __restrict__ bias,
            bf16* __restrict__ Y, int ldy,
            int K, int doRelu) {
  __shared__ unsigned int sWu[32 * 260];   // up to 32 rows x 520 bf16
  const int SLD = 520;
  const int NR  = 16 * NT;

  const int m0  = blockIdx.x * 64;
  const int n0  = blockIdx.y * NR;
  const int tid = threadIdx.x;

  // cooperative stage of NR weight rows into LDS (packed 2x bf16 words)
  const int Kw = K >> 1;
  for (int w = tid; w < NR * Kw; w += 128) {
    const int n  = w / Kw;
    const int k2 = w - n * Kw;
    sWu[n * 260 + k2] = ((const unsigned int*)(W + (size_t)(n0 + n) * ldw))[k2];
  }
  __syncthreads();

  const int wave = tid >> 5;
  const int lane = tid & 31;
  const int m    = m0 + wave * 16;
  const bf16* sW = (const bf16*)sWu;

  v8f acc[NT];
#pragma unroll
  for (int t = 0; t < NT; ++t) acc[t] = v8f{};

  for (int kb = 0; kb < K; kb += 32) {
    v16bf a = load_a_frag(X + (size_t)m * ldx, ldx, kb, lane);
#pragma unroll
    for (int t = 0; t < NT; ++t) {
      v16bf b = load_b_frag(sW + (size_t)t * 16 * SLD, SLD, kb, lane);
      acc[t] = wmma_bf16(a, b, acc[t]);
    }
  }

  // epilogue: D layout -> lane&15 = n, VGPR r = row r + 8*(lane>>4)
  const int hoff = 8 * (lane >> 4);
#pragma unroll
  for (int t = 0; t < NT; ++t) {
    const int n    = n0 + t * 16 + (lane & 15);
    const float bn = bias[n];
#pragma unroll
    for (int r = 0; r < 8; ++r) {
      float v = acc[t][r] + bn;
      if (doRelu) v = v > 0.0f ? v : 0.0f;
      Y[(size_t)(m + r + hoff) * ldy + n] = (bf16)v;
    }
  }
}

// ----- PatchNCE: per-wave 16 query rows, sweep 256 key tiles with WMMA.
// Query fragments are hoisted out of the sweep. Each lane keeps a private
// per-row (max, sum) over its own columns; one cross-lane logsumexp merge at
// the end. Diagonal of the unmasked score row == l_pos, so masking cancels:
//   loss = mean( LSE_t(l_neg/tau) - l_pos/tau ).
template <int CP>
__global__ void __launch_bounds__(128)
k_nce_wmma(const bf16* __restrict__ Fq, const bf16* __restrict__ Fk,
           float invTau, float scale, float* __restrict__ out) {
  const int wave = threadIdx.x >> 5;
  const int lane = threadIdx.x & 31;
  const int qb   = blockIdx.x * 4 + wave;        // 0..1023
  const int b    = qb >> 8;                      // batch
  const int qloc = qb & 255;                     // qblock within batch
  const int half = lane >> 4;

  const bf16* fqBase  = Fq + ((size_t)b * S_PAIRS + (size_t)qloc * 16) * CP;
  const bf16* fkBatch = Fk + (size_t)b * S_PAIRS * CP;

  constexpr int KS = CP / 32;
  v16bf aFrag[KS];
#pragma unroll
  for (int kk = 0; kk < KS; ++kk)
    aFrag[kk] = load_a_frag(fqBase, CP, kk * 32, lane);

  float lmax[8], lsum[8], lpos[8];
#pragma unroll
  for (int r = 0; r < 8; ++r) { lmax[r] = -3.0e38f; lsum[r] = 0.0f; lpos[r] = 0.0f; }

  for (int kt = 0; kt < S_PAIRS / 16; ++kt) {
    const bf16* fkTile = fkBatch + (size_t)kt * 16 * CP;
    v8f acc = {};
#pragma unroll
    for (int kk = 0; kk < KS; ++kk)
      acc = wmma_bf16(aFrag[kk], load_b_frag(fkTile, CP, kk * 32, lane), acc);

    const bool diagTile = (kt == qloc);
#pragma unroll
    for (int r = 0; r < 8; ++r) {
      const float s = acc[r] * invTau;           // logits / tau
      if (diagTile) {
        // diag element of row m sits at group-lane m (l_pos == l_neg[s,s])
        lpos[r] = __shfl(s, (half << 4) + (r + 8 * half), 32);
      }
      const float nm = fmaxf(lmax[r], s);        // lane-private online LSE
      lsum[r] = lsum[r] * __expf(lmax[r] - nm) + __expf(s - nm);
      lmax[r] = nm;
    }
  }

  // merge lane-private (max, sum) across the 16-lane group holding each row
#pragma unroll
  for (int r = 0; r < 8; ++r) {
    float m = lmax[r], su = lsum[r];
#pragma unroll
    for (int d = 1; d < 16; d <<= 1) {
      const float om = __shfl_xor(m, d, 32);
      const float os = __shfl_xor(su, d, 32);
      const float nm = fmaxf(m, om);
      su = su * __expf(m - nm) + os * __expf(om - nm);
      m = nm;
    }
    lmax[r] = m; lsum[r] = su;
  }

  float part = 0.0f;
#pragma unroll
  for (int r = 0; r < 8; ++r)
    part += (lmax[r] + __logf(lsum[r])) - lpos[r];
  if (lane & 15) part = 0.0f;                    // one lane per half keeps 8 rows
  part += __shfl_xor(part, 16, 32);
  if (lane == 0) unsafeAtomicAdd(out, part * scale);
}

// ---------------------------------------------------------------------------
extern "C" void kernel_launch(void* const* d_in, const int* in_sizes, int n_in,
                              void* d_out, int out_size, void* d_ws, size_t ws_size,
                              hipStream_t stream) {
  (void)in_sizes; (void)n_in; (void)ws_size;
  float* out = (float*)d_out;
  k_zero_f32<<<1, 64, 0, stream>>>(out, out_size);

  char* ws = (char*)d_ws;
  bf16* bufA = (bf16*)(ws);                         // 16 MB ping
  bf16* bufB = (bf16*)(ws + (size_t)(16 << 20));    // 16 MB pong
  bf16* Fq   = (bf16*)(ws + (size_t)(32 << 20));    //  4 MB
  bf16* Fk   = (bf16*)(ws + (size_t)(36 << 20));    //  4 MB
  bf16* Wb0  = (bf16*)(ws + (size_t)(40 << 20));
  bf16* Wb1  = (bf16*)(ws + (size_t)(40 << 20) + (512 << 10));
  bf16* Wb2  = (bf16*)(ws + (size_t)(40 << 20) + (1 << 20));

  static const int Cs[4]  = {64, 128, 256, 512};
  static const int HWs[4] = {256 * 256, 128 * 128, 64 * 64, 32 * 32};
  const float invTau = 100.0f;                      // 1 / 0.01
  const float scale  = 1.0f / (float)MROWS;         // mean over [B, S]

  for (int i = 0; i < 4; ++i) {
    const int base = i * 10;
    const float* fq  = (const float*)d_in[base + 0];
    const float* fk  = (const float*)d_in[base + 1];
    const int*   cid = (const int*)  d_in[base + 2];
    const int*   nid = (const int*)  d_in[base + 3];
    const float* w0  = (const float*)d_in[base + 4];
    const float* b0  = (const float*)d_in[base + 5];
    const float* w1  = (const float*)d_in[base + 6];
    const float* b1  = (const float*)d_in[base + 7];
    const float* w2  = (const float*)d_in[base + 8];
    const float* b2  = (const float*)d_in[base + 9];

    const int C    = Cs[i];
    const int HW   = HWs[i];
    const int Cout = C / 4;
    const int Cpad = Cout < 32 ? 32 : Cout;

    // weights -> bf16
    k_cvt_bf16<<<(C * C + 255) / 256, 256, 0, stream>>>(w0, Wb0, C * C);
    k_cvt_bf16<<<(C * C + 255) / 256, 256, 0, stream>>>(w1, Wb1, C * C);
    k_cvt_bf16<<<(Cout * C + 255) / 256, 256, 0, stream>>>(w2, Wb2, Cout * C);

    if (Cpad != Cout) {                              // zero K-padding columns
      const int nw = MROWS * Cpad / 2;
      k_zero_u32<<<(nw + 255) / 256, 256, 0, stream>>>((unsigned int*)Fq, nw);
      k_zero_u32<<<(nw + 255) / 256, 256, 0, stream>>>((unsigned int*)Fk, nw);
    }

    const float* feats[2] = {fq, fk};
    bf16*        Fs[2]    = {Fq, Fk};
    for (int t = 0; t < 2; ++t) {
      const int total = MROWS * C;
      k_gather_diff<<<(total + 255) / 256, 256, 0, stream>>>(feats[t], cid, nid,
                                                             bufA, C, HW);
      dim3 gridG(MROWS / 64, C / 32);
      k_gemm_wmma<2><<<gridG, 128, 0, stream>>>(bufA, C, Wb0, C, b0, bufB, C, C, 1);
      k_gemm_wmma<2><<<gridG, 128, 0, stream>>>(bufB, C, Wb1, C, b1, bufA, C, C, 1);
      if (Cout >= 32) {
        dim3 gridO(MROWS / 64, Cout / 32);
        k_gemm_wmma<2><<<gridO, 128, 0, stream>>>(bufA, C, Wb2, C, b2, Fs[t], Cpad, C, 0);
      } else {
        dim3 gridO(MROWS / 64, Cout / 16);
        k_gemm_wmma<1><<<gridO, 128, 0, stream>>>(bufA, C, Wb2, C, b2, Fs[t], Cpad, C, 0);
      }
    }

    switch (Cpad) {
      case 32:
        k_nce_wmma<32><<<256, 128, 0, stream>>>(Fq, Fk, invTau, scale, out);
        break;
      case 64:
        k_nce_wmma<64><<<256, 128, 0, stream>>>(Fq, Fk, invTau, scale, out);
        break;
      default:
        k_nce_wmma<128><<<256, 128, 0, stream>>>(Fq, Fk, invTau, scale, out);
        break;
    }
  }
}
